// MSAOverflowBuffer_29386166239831
// MI455X (gfx1250) — matrix operand order repaired and checked by reading, not compiled, not run on hardware
//
#include <hip/hip_runtime.h>
#include <hip/hip_bf16.h>
#include <math.h>

// ---------------------------------------------------------------------------
// MI455X plan:
//  * 3 large GEMMs (qr, kr, scores) + 1 small GEMM (out):
//      v_wmma_f32_16x16x32_bf16; block tile 128M x 128N (8 waves as 4Mx2N,
//      wave tile 32x64 -> 8 WMMA per wave per K-step); K-steps of 32 with
//      A/B panels staged to LDS by the Tensor Data Mover (TENSORcnt,
//      double-buffered) so the WMMA stream never waits on global latency.
//  * f32->bf16 conversion and per-head L2 norm: bandwidth-trivial passes.
//  * top-16 + softmax + blend: one 256-thread block per row.
// ---------------------------------------------------------------------------

typedef __attribute__((ext_vector_type(16))) __bf16        v16bf;
typedef __attribute__((ext_vector_type(8)))  float         v8f;
typedef __attribute__((ext_vector_type(4)))  unsigned int  v4u;
typedef __attribute__((ext_vector_type(8)))  int           v8i_t;
typedef __attribute__((ext_vector_type(4)))  int           v4i_t;

union FragBF { uint4 u[2]; v16bf v; };

#define DIMD 1024
#define PCNT 32768
#define BCNT 1024

// ---------------- f32 -> bf16 conversion ----------------
__global__ void __launch_bounds__(256)
cvt_f32_bf16(const float* __restrict__ in, __hip_bfloat16* __restrict__ out, size_t n) {
    size_t i = (size_t)blockIdx.x * 256 + threadIdx.x;
    size_t stride = (size_t)gridDim.x * 256;
    for (; i < n; i += stride) out[i] = __float2bfloat16(in[i]);
}

// ---------------- per-(row,head) L2 normalize, emit bf16 ----------------
__global__ void __launch_bounds__(256)
headnorm_bf16(const float* __restrict__ in, __hip_bfloat16* __restrict__ out) {
    const int seg = blockIdx.x;            // (row * H + head), hd == 256
    const int t   = threadIdx.x;
    const float x = in[(size_t)seg * 256 + t];
    __shared__ float red[256];
    red[t] = x * x;
    __syncthreads();
    for (int s = 128; s > 0; s >>= 1) {
        if (t < s) red[t] += red[t + s];
        __syncthreads();
    }
    const float nrm = sqrtf(red[0]);
    const float den = fmaxf(nrm, 1e-12f);
    out[(size_t)seg * 256 + t] = __float2bfloat16(x / den);
}

// ---------------- TDM: async 2-D bf16 tile (rows x 32) global -> LDS --------
// D# per CDNA5 ISA: group0 = {count/type, lds_addr,
// global_addr}, group1 = {data_size, tensor dims/strides, tile dims}.
// tile_dim0 = 32 contiguous K elements, tile_dim1 = rows, stride = K elems.
__device__ __forceinline__ void tdm_load_tile_bf16(unsigned lds_byte_off,
                                                   const void* gaddr,
                                                   unsigned rows,
                                                   unsigned row_stride_elems) {
    unsigned long long ga = (unsigned long long)gaddr;
    v4u g0;
    g0[0] = 1u;                                           // count=1 (valid user D#)
    g0[1] = lds_byte_off;                                 // lds_addr [63:32]
    g0[2] = (unsigned)(ga & 0xFFFFFFFFu);                 // global_addr lo
    g0[3] = (unsigned)((ga >> 32) & 0x01FFFFFFu)          // global_addr hi [120:96]
          | 0x80000000u;                                  // type=2 ("image") [127:126]
    v8i_t g1;
    g1[0] = 0x00010000;                   // data_size=1 (2 bytes), mask=0
    g1[1] = (int)(32u << 16);             // tensor_dim0 = 32 (lo16 @ [63:48])
    g1[2] = (int)(rows << 16);            // tensor_dim0 hi=0 | tensor_dim1 lo16
    g1[3] = (int)(32u << 16);             // tensor_dim1 hi=0 | tile_dim0 = 32
    g1[4] = (int)rows;                    // tile_dim1 = rows, tile_dim2 = 0
    g1[5] = (int)row_stride_elems;        // tensor_dim0_stride lo32
    g1[6] = 0;                            // stride hi16 | tensor_dim1_stride lo
    g1[7] = 0;
    v4i_t z4 = {0, 0, 0, 0};
#if __clang_major__ >= 23
    v8i_t z8 = {0, 0, 0, 0, 0, 0, 0, 0};
    __builtin_amdgcn_tensor_load_to_lds(g0, g1, z4, z4, z8, 0);
#else
    __builtin_amdgcn_tensor_load_to_lds(g0, g1, z4, z4, 0);
#endif
}

// ---------------- C[M,N] = alpha * A[M,K] @ B[N,K]^T  (bf16 in, f32 out) ----
// Block: 8 waves (4Mx2N) -> 128 rows x 128 cols; wave tile 32x64 (8 WMMA/step).
// TDM double-buffers A (128x32) and B (128x32) panels; fragments via
// ds_read_b128 per ISA layouts:
//   A 16x32: lane<16 row=lane K{0..7,16..23}; lane>=16 row=lane-16 K{8..15,24..31}
//   B 32x16: lane col = lane%16, K same as A: lane<16 K{0..7,16..23}; lane>=16 K{8..15,24..31}
// LDS map (dynamic, 32 KB): A buf0 @0, A buf1 @8192, B buf0 @16384, B buf1 @24576
__global__ void __launch_bounds__(256)
gemm_bf16_nt(const __hip_bfloat16* __restrict__ A,
             const __hip_bfloat16* __restrict__ Bw,
             float* __restrict__ C,
             int M, int N, int K, float alpha) {
    extern __shared__ char smem[];
    const int lane   = threadIdx.x & 31;
    const int wave   = threadIdx.x >> 5;
    const int hi     = lane >> 4;
    const int lr     = lane & 15;
    const int wave_m = wave & 3;        // 4 waves over M (32 rows each)
    const int wave_n = wave >> 2;       // 2 waves over N (64 cols each)
    const int ntiles = N >> 7;
    const int tile_n = blockIdx.x % ntiles;
    const int tile_m = blockIdx.x / ntiles;
    const int row0   = tile_m << 7;     // 128-row block tile
    const int col0   = tile_n << 7;     // 128-col block tile
    const bool issuer = (wave == 0);    // wave-uniform; TDM issued once per WG
    const int nk = K >> 5;

    if (issuer) {
        tdm_load_tile_bf16(0,     A  + (size_t)row0 * K, 128, (unsigned)K);
        tdm_load_tile_bf16(16384, Bw + (size_t)col0 * K, 128, (unsigned)K);
        if (nk > 1) {
            tdm_load_tile_bf16(8192,  A  + (size_t)row0 * K + 32, 128, (unsigned)K);
            tdm_load_tile_bf16(24576, Bw + (size_t)col0 * K + 32, 128, (unsigned)K);
        }
    }

    v8f acc[2][4] = {};
    const int arow_l = (wave_m << 5) + lr;   // + mi*16
    const int bcol_l = (wave_n << 6) + lr;   // + nt*16

    for (int s = 0; s < nk; ++s) {
        const int cur = s & 1;
        if (issuer) {
            if (s + 1 < nk) __builtin_amdgcn_s_wait_tensorcnt(2);  // step s done
            else            __builtin_amdgcn_s_wait_tensorcnt(0);  // drain
        }
        __syncthreads();    // LDS panels for step s visible to all waves

        const char* Ab = smem + cur * 8192;
        const char* Bb = smem + 16384 + cur * 8192;

        FragBF fa[2];
#pragma unroll
        for (int mi = 0; mi < 2; ++mi) {
            const char* pa = Ab + (arow_l + mi * 16) * 64 + hi * 16;
            fa[mi].u[0] = *(const uint4*)(pa);
            fa[mi].u[1] = *(const uint4*)(pa + 32);
        }
#pragma unroll
        for (int nt = 0; nt < 4; ++nt) {
            FragBF fb;
            const char* pb = Bb + (bcol_l + nt * 16) * 64 + hi * 16;
            fb.u[0] = *(const uint4*)(pb);
            fb.u[1] = *(const uint4*)(pb + 32);
#pragma unroll
            for (int mi = 0; mi < 2; ++mi) {
                acc[mi][nt] = __builtin_amdgcn_wmma_f32_16x16x32_bf16(
                    false, fa[mi].v, false, fb.v, (short)0, acc[mi][nt], false, false);
            }
        }

        __syncthreads();    // all waves done reading buf[cur]
        if (issuer && (s + 2 < nk)) {
            const int kb2 = (s + 2) << 5;
            tdm_load_tile_bf16(cur * 8192,         A  + (size_t)row0 * K + kb2, 128, (unsigned)K);
            tdm_load_tile_bf16(16384 + cur * 8192, Bw + (size_t)col0 * K + kb2, 128, (unsigned)K);
        }
    }

#pragma unroll
    for (int mi = 0; mi < 2; ++mi) {
#pragma unroll
        for (int nt = 0; nt < 4; ++nt) {
            const int cn = col0 + (wave_n << 6) + nt * 16 + lr;
#pragma unroll
            for (int r = 0; r < 8; ++r) {
                const int cm = row0 + (wave_m << 5) + mi * 16 + hi * 8 + r;
                C[(size_t)cm * N + cn] = acc[mi][nt][r] * alpha;
            }
        }
    }
}

// ---------------- top-16 + softmax + prototype blend (per row) --------------
__global__ void __launch_bounds__(256)
topk_blend(const float* __restrict__ scores,
           const float* __restrict__ protos,
           __hip_bfloat16* __restrict__ retr) {
    const int b = blockIdx.x;
    const int t = threadIdx.x;

    float v[16]; int id[16];
#pragma unroll
    for (int j = 0; j < 16; ++j) { v[j] = -3.0e38f; id[j] = 0; }
    const float* row = scores + (size_t)b * PCNT;
    for (int i = 0; i < PCNT / 256; ++i) {
        const int p = t + i * 256;           // coalesced
        const float s = row[p];
        if (s > v[15]) {
            v[15] = s; id[15] = p;
#pragma unroll
            for (int j = 15; j > 0; --j) {
                if (v[j] > v[j - 1]) {
                    float tv = v[j]; v[j] = v[j - 1]; v[j - 1] = tv;
                    int   ti = id[j]; id[j] = id[j - 1]; id[j - 1] = ti;
                }
            }
        }
    }

    __shared__ float          sv[2][256 * 16];   // 32 KB
    __shared__ unsigned short si[2][256 * 16];   // 16 KB (P fits in u16)
#pragma unroll
    for (int j = 0; j < 16; ++j) {
        sv[0][t * 16 + j] = v[j];
        si[0][t * 16 + j] = (unsigned short)id[j];
    }
    __syncthreads();

    int src = 0;
    for (int stride = 1; stride < 256; stride <<= 1) {
        const int dst = src ^ 1;
        if ((t & (2 * stride - 1)) == 0) {
            const int a = t * 16, c = (t + stride) * 16;
            int i = 0, j = 0;
            for (int r = 0; r < 16; ++r) {
                const float va = sv[src][a + i], vb = sv[src][c + j];
                if (va >= vb) { sv[dst][a + r] = va; si[dst][a + r] = si[src][a + i]; ++i; }
                else          { sv[dst][a + r] = vb; si[dst][a + r] = si[src][c + j]; ++j; }
            }
        }
        __syncthreads();
        src = dst;
    }

    __shared__ float wsm[16];
    __shared__ int   sidx[16];
    if (t == 0) {
        const float smax = sv[src][0];   // sorted desc
        float e[16], sum = 0.f;
#pragma unroll
        for (int k = 0; k < 16; ++k) { e[k] = __expf(sv[src][k] - smax); sum += e[k]; }
#pragma unroll
        for (int k = 0; k < 16; ++k) { wsm[k] = e[k] / sum; sidx[k] = si[src][k]; }
    }
    __syncthreads();

#pragma unroll
    for (int dd = 0; dd < 4; ++dd) {
        const int d = t + dd * 256;
        float acc = 0.f;
        for (int k = 0; k < 16; ++k)
            acc += wsm[k] * protos[(size_t)sidx[k] * DIMD + d];
        retr[(size_t)b * DIMD + d] = __float2bfloat16(acc);
    }
}

// ---------------------------------------------------------------------------
extern "C" void kernel_launch(void* const* d_in, const int* in_sizes, int n_in,
                              void* d_out, int out_size, void* d_ws, size_t ws_size,
                              hipStream_t stream) {
    const float* h      = (const float*)d_in[0];   // [B, D]
    const float* protos = (const float*)d_in[1];   // [P, D]
    const float* Wqr    = (const float*)d_in[2];   // [D, D]
    const float* Wkr    = (const float*)d_in[3];   // [D, D]
    const float* Wout   = (const float*)d_in[4];   // [D, D]
    float*       out    = (float*)d_out;           // [B, D]

    const int B = BCNT, D = DIMD, P = PCNT;

    char* ws = (char*)d_ws;
    size_t off = 0;
    auto carve = [&](size_t bytes) -> char* {
        char* p = ws + off;
        off += (bytes + 255) & ~(size_t)255;
        return p;
    };

    __hip_bfloat16* hbf    = (__hip_bfloat16*)carve((size_t)B * D * 2);  //   2 MB
    __hip_bfloat16* pbf    = (__hip_bfloat16*)carve((size_t)P * D * 2);  //  64 MB
    __hip_bfloat16* wqrbf  = (__hip_bfloat16*)carve((size_t)D * D * 2);  //   2 MB
    __hip_bfloat16* wkrbf  = (__hip_bfloat16*)carve((size_t)D * D * 2);  //   2 MB
    __hip_bfloat16* woutbf = (__hip_bfloat16*)carve((size_t)D * D * 2);  //   2 MB
    float*          qr32   = (float*)carve((size_t)B * D * 4);           //   4 MB
    __hip_bfloat16* qnbf   = (__hip_bfloat16*)carve((size_t)B * D * 2);  //   2 MB
    __hip_bfloat16* knbf   = (__hip_bfloat16*)carve((size_t)P * D * 2);  //  64 MB
    __hip_bfloat16* rbf    = (__hip_bfloat16*)carve((size_t)B * D * 2);  //   2 MB
    float*          kr32   = (float*)carve((size_t)P * D * 4);           // 128 MB
    float*          sc32   = kr32;  // alias: kr dead once knbf is built

    const size_t SMEM = 32768;      // 2 x (128x32 A + 128x32 B) bf16 panels

    // 1) one-time bf16 conversions
    cvt_f32_bf16<<<(B * D + 255) / 256, 256, 0, stream>>>(h, hbf, (size_t)B * D);
    cvt_f32_bf16<<<((size_t)P * D + 255) / 256, 256, 0, stream>>>(protos, pbf, (size_t)P * D);
    cvt_f32_bf16<<<(D * D + 255) / 256, 256, 0, stream>>>(Wqr, wqrbf, (size_t)D * D);
    cvt_f32_bf16<<<(D * D + 255) / 256, 256, 0, stream>>>(Wkr, wkrbf, (size_t)D * D);
    cvt_f32_bf16<<<(D * D + 255) / 256, 256, 0, stream>>>(Wout, woutbf, (size_t)D * D);

    // 2) qr = h @ Wqr^T
    gemm_bf16_nt<<<(B / 128) * (D / 128), 256, SMEM, stream>>>(hbf, wqrbf, qr32, B, D, D, 1.0f);
    // 3) per-head L2 norm -> qn
    headnorm_bf16<<<B * 4, 256, 0, stream>>>(qr32, qnbf);

    // 4) kr = protos @ Wkr^T
    gemm_bf16_nt<<<(P / 128) * (D / 128), 256, SMEM, stream>>>(pbf, wkrbf, kr32, P, D, D, 1.0f);
    // 5) per-head L2 norm -> kn
    headnorm_bf16<<<P * 4, 256, 0, stream>>>(kr32, knbf);

    // 6) scores = 2.5 * qn @ kn^T   (1/(H*T) = 2.5)
    gemm_bf16_nt<<<(B / 128) * (P / 128), 256, SMEM, stream>>>(qnbf, knbf, sc32, B, P, D, 2.5f);

    // 7) top-16 + softmax + prototype blend
    topk_blend<<<B, 256, 0, stream>>>(sc32, protos, rbf);

    // 8) out = retrieved @ Wout^T
    gemm_bf16_nt<<<(B / 128) * (D / 128), 256, SMEM, stream>>>(rbf, woutbf, out, B, D, D, 1.0f);
}
